// Conv_DCNv3_29652454212242
// MI455X (gfx1250) — matrix operand order repaired and checked by reading, not compiled, not run on hardware
//
#include <hip/hip_runtime.h>
#include <hip/hip_bf16.h>
#include <math.h>

// ---------------------------------------------------------------------------
// DCNv3 block for MI455X (gfx1250, wave32, WMMA + TDM).
// Pipeline is bandwidth-bound => f16 WMMA GEMMs (f32 accumulate), NHWC
// activation layout for coalesced deformable gathers, Tensor Data Mover
// for GEMM A-operand staging (global -> LDS DMA, TENSORcnt tracked).
// ---------------------------------------------------------------------------

typedef __attribute__((ext_vector_type(16))) _Float16     v16h;
typedef __attribute__((ext_vector_type(8)))  _Float16     v8h;
typedef __attribute__((ext_vector_type(8)))  float        v8f;
typedef __attribute__((ext_vector_type(4)))  unsigned int v4u;
typedef __attribute__((ext_vector_type(8)))  int          v8i;
typedef __attribute__((ext_vector_type(4)))  int          v4i;

#define NB    4
#define C_    256
#define H_    64
#define W_    64
#define HW_   4096
#define MTOT  16384          // N*H*W rows
#define G_    8
#define GC_   32
#define P_    9
#define NOFF  144            // G*P*2
#define NMSK  72             // G*P
#define NOM   224            // 144 + 72 padded to multiple of 32
#define LDA_  264            // A slab LDS row stride in halves (256 + 8 pad)
#define LDB_  264            // B slab LDS col stride in halves

// ---------------------------------------------------------------------------
// 0) Parameter prep: f32 -> f16 weights; concat+pad offset/mask weights.
// ---------------------------------------------------------------------------
__global__ void __launch_bounds__(256)
prep_params(const float* __restrict__ w_in, const float* __restrict__ w_off,
            const float* __restrict__ w_mask, const float* __restrict__ b_off,
            const float* __restrict__ b_mask, const float* __restrict__ w_out,
            _Float16* __restrict__ Win, _Float16* __restrict__ Wom,
            _Float16* __restrict__ Wout, float* __restrict__ bom)
{
    int i = blockIdx.x * 256 + threadIdx.x;           // 0..65535
    if (i < C_ * C_) {
        Win[i]  = (_Float16)w_in[i];
        Wout[i] = (_Float16)w_out[i];
    }
    if (i < C_ * NOM) {
        int k = i / NOM, n = i % NOM;
        float v = 0.f;
        if (n < NOFF)              v = w_off[k * NOFF + n];
        else if (n < NOFF + NMSK)  v = w_mask[k * NMSK + (n - NOFF)];
        Wom[i] = (_Float16)v;
    }
    if (i < NOM) {
        float v = 0.f;
        if (i < NOFF)              v = b_off[i];
        else if (i < NOFF + NMSK)  v = b_mask[i - NOFF];
        bom[i] = v;
    }
}

// ---------------------------------------------------------------------------
// 1) NCHW f32 -> NHWC f16 via LDS tile transpose (coalesced both sides).
//    Block: 32 pixels x 256 channels.
// ---------------------------------------------------------------------------
__global__ void __launch_bounds__(256)
nchw_to_nhwc_f16(const float* __restrict__ x, _Float16* __restrict__ A0)
{
    __shared__ float lt[256][33];
    const int t    = threadIdx.x;
    const int nimg = blockIdx.x >> 7;                 // 4 images * 128 tiles
    const int hw0  = (blockIdx.x & 127) * 32;
    #pragma unroll 4
    for (int c0 = 0; c0 < 256; c0 += 8) {
        int c = c0 + (t >> 5);
        lt[c][t & 31] = x[((size_t)(nimg * C_ + c) << 12) + hw0 + (t & 31)];
    }
    __syncthreads();
    #pragma unroll 4
    for (int pix = 0; pix < 32; pix++) {
        A0[((size_t)((nimg << 12) + hw0 + pix)) * C_ + t] = (_Float16)lt[t][pix];
    }
}

// ---------------------------------------------------------------------------
// 2) WMMA GEMM: C[M x Nc] = A[M x 256] * B[256 x Nc] + bias, f16 in f32 acc.
//    Block = 8 waves, 64(M) x 32(N) macro tile, 1 wave = one 16x16 tile.
//
//    A slab (64 x 256 f16) is DMA'd into LDS by the Tensor Data Mover via a
//    single tensor_load_to_lds descriptor. The D# pad fields insert 16B of
//    padding after every 512B row => LDS row stride 264 halves (16B aligned,
//    bank-staggered) -- same layout a manual padded copy would produce, but
//    with zero VALU/VGPR cost and tracked by TENSORcnt.
//    B slab (256 x 32) is staged transposed by all 8 waves once.
//    Inner loop is then pure ds_load_b128 fragment fetch + v_wmma, no
//    barriers, matching the ISA f16 operand layouts:
//      A 16x32: lane L(<16): K {0..7,16..23}; lane L+16: K {8..15,24..31}
//      B 32x16: lanes 0-15 cols, K {0..7,16..23}; lanes 16-31 cols, K {8..15,24..31}
//    EPI 0: f32 row-major store.  EPI 1: +b_out, BN, SiLU, NCHW f32 store.
// ---------------------------------------------------------------------------
template <int EPI>
__global__ void __launch_bounds__(256)
gemm_wmma(const _Float16* __restrict__ A, const _Float16* __restrict__ B,
          const float* __restrict__ bias, float* __restrict__ out,
          int M, int Nc,
          const float* __restrict__ bn_g, const float* __restrict__ bn_b,
          const float* __restrict__ bn_mean, const float* __restrict__ bn_var)
{
    __shared__ _Float16 la [64][LDA_];   // A slab, TDM-padded rows
    __shared__ _Float16 lbt[32][LDB_];   // B slab transposed: [col][k]
    const int t    = threadIdx.x;
    const int lane = t & 31;
    const int wv   = t >> 5;
    const int tr   = wv & 3;             // wave's 16-row tile within 64
    const int tc   = wv >> 2;            // wave's 16-col tile within 32
    const int m0   = blockIdx.y * 64;
    const int n0   = blockIdx.x * 32;
    (void)M;

    // ---- A slab: one TDM descriptor, issued by wave 0 (EXEC-independent) ---
    if (wv == 0) {
        unsigned           lds_addr = (unsigned)(uintptr_t)&la[0][0];
        unsigned long long ga = (unsigned long long)(uintptr_t)(A + (size_t)m0 * 256);
        // D# group 0: count=1 | lds_addr | global_addr | type=2
        v4u g0 = { 1u, lds_addr, (unsigned)ga,
                   (unsigned)(ga >> 32) | 0x80000000u };
        // D# group 1: data_size=2B(1), pad_enable, pad_interval=6 (128 DW =
        // one 512B row), pad_amount=3 (4 DW = 16B); tensor dims 0xFFFF (tiles
        // never OOB); tile = 256 x 64; dim0 stride = 256 elements.
        v8i g1 = { (int)((1u << 16) | (1u << 20) | (6u << 22) | (3u << 25)),
                   (int)0xFFFF0000u,      // tensor_dim0[15:0] in bits 63:48
                   (int)0xFFFF0000u,      // tensor_dim1[15:0] in bits 95:80
                   (int)(256u << 16),     // tile_dim0 = 256 (bits 127:112)
                   64,                    // tile_dim1 = 64, tile_dim2 = 0
                   256,                   // tensor_dim0_stride (bits 191:160)
                   0, 0 };
        v4i gz4 = { 0, 0, 0, 0 };
        v8i gz8 = { 0, 0, 0, 0, 0, 0, 0, 0 };
        __builtin_amdgcn_tensor_load_to_lds(g0, g1, gz4, gz4, gz8, 0);
        __builtin_amdgcn_s_wait_tensorcnt((short)0);   // s_wait_tensorcnt 0
    }

    // ---- B slab: 256x32 halves staged transposed by all threads -----------
    {
        const int seg = t & 7;
        #pragma unroll
        for (int it = 0; it < 8; it++) {
            int kr = it * 32 + (t >> 3);
            uint2 v = *(const uint2*)(B + (size_t)kr * Nc + n0 + seg * 4);
            const _Float16* hv = (const _Float16*)&v;
            #pragma unroll
            for (int q = 0; q < 4; q++) lbt[seg * 4 + q][kr] = hv[q];
        }
    }
    __syncthreads();

    // ---- Inner loop: barrier-free ds_load_b128 + v_wmma --------------------
    const int arow = tr * 16 + (lane & 15);
    const int aseg = (lane >> 4) * 8;
    const int bcol = tc * 16 + (lane & 15);
    const int brow = (lane >> 4) * 8;

    v8f acc = {};
    #pragma unroll
    for (int k0 = 0; k0 < 256; k0 += 32) {
        v8h a_lo = *(const v8h*)&la[arow][k0 + aseg];
        v8h a_hi = *(const v8h*)&la[arow][k0 + 16 + aseg];
        v16h afrag = __builtin_shufflevector(a_lo, a_hi,
            0, 1, 2, 3, 4, 5, 6, 7, 8, 9, 10, 11, 12, 13, 14, 15);

        v8h b_lo = *(const v8h*)&lbt[bcol][k0 + brow];
        v8h b_hi = *(const v8h*)&lbt[bcol][k0 + 16 + brow];
        v16h bfrag = __builtin_shufflevector(b_lo, b_hi,
            0, 1, 2, 3, 4, 5, 6, 7, 8, 9, 10, 11, 12, 13, 14, 15);

        acc = __builtin_amdgcn_wmma_f32_16x16x32_f16(
            /*neg_a=*/false, afrag, /*neg_b=*/false, bfrag,
            /*c_mod=*/(short)0, acc, /*reuse_a=*/false, /*reuse_b=*/false);
    }

    // Epilogue. C/D layout: lanes 0-15: VGPR r -> M=r, N=lane;
    //                       lanes 16-31: VGPR r -> M=r+8, N=lane-16.
    const int ncol = n0 + tc * 16 + (lane & 15);
    const float bv = bias ? bias[ncol] : 0.f;
    #pragma unroll
    for (int r = 0; r < 8; r++) {
        int mrow = m0 + tr * 16 + r + 8 * (lane >> 4);
        float v = acc[r] + bv;
        if constexpr (EPI == 0) {
            out[(size_t)mrow * Nc + ncol] = v;
        } else {
            const int c = ncol;
            v = (v - bn_mean[c]) * rsqrtf(bn_var[c] + 1e-5f) * bn_g[c] + bn_b[c];
            float y = v / (1.f + __expf(-v));                   // SiLU
            int nimg = mrow >> 12, hw = mrow & 4095;
            out[((size_t)(nimg * C_ + c) << 12) + hw] = y;      // NCHW
        }
    }
}

// ---------------------------------------------------------------------------
// 3) Depthwise 3x3 + bias + LayerNorm(C) + exact GELU. Block = one pixel,
//    thread = channel; reads NHWC f16 (channel-contiguous, coalesced).
// ---------------------------------------------------------------------------
__global__ void __launch_bounds__(256)
dw_ln_gelu(const _Float16* __restrict__ A0, const float* __restrict__ w_dw,
           const float* __restrict__ b_dw, const float* __restrict__ ln_g,
           const float* __restrict__ ln_b, _Float16* __restrict__ A1)
{
    __shared__ float red[256];
    const int m = blockIdx.x;
    const int c = threadIdx.x;
    const int nimg = m >> 12, hw = m & 4095, h = hw >> 6, w = hw & 63;

    float v = b_dw[c];
    #pragma unroll
    for (int dy = 0; dy < 3; dy++) {
        #pragma unroll
        for (int dx = 0; dx < 3; dx++) {
            int hh = h + dy - 1, ww = w + dx - 1;
            if (hh >= 0 && hh < H_ && ww >= 0 && ww < W_)
                v += (float)A0[((size_t)((nimg << 12) + (hh << 6) + ww)) * C_ + c]
                     * w_dw[c * 9 + dy * 3 + dx];
        }
    }
    red[c] = v; __syncthreads();
    for (int s = 128; s > 0; s >>= 1) { if (c < s) red[c] += red[c + s]; __syncthreads(); }
    float mean = red[0] * (1.f / 256.f);
    __syncthreads();
    float d = v - mean;
    red[c] = d * d; __syncthreads();
    for (int s = 128; s > 0; s >>= 1) { if (c < s) red[c] += red[c + s]; __syncthreads(); }
    float var = red[0] * (1.f / 256.f);

    float xn = d * rsqrtf(var + 1e-5f) * ln_g[c] + ln_b[c];
    float ge = 0.5f * xn * (1.f + erff(xn * 0.70710678118654752f));   // exact GELU
    A1[(size_t)m * C_ + c] = (_Float16)ge;
}

// ---------------------------------------------------------------------------
// 4) Deformable sampling. One wave per (pixel, group); lane = channel within
//    group => every bilinear tap is one coalesced 128B read from NHWC xproj.
//    Softmax over P=9 done in registers (offsets/logits broadcast per wave).
//    Padding (PAD=1) folded into tap bounds: padded coord t valid iff 1..64.
// ---------------------------------------------------------------------------
__device__ __forceinline__ float dcn_tap(const float* __restrict__ xp,
                                         int nimg, int y, int x, int c)
{
    if (x < 1 || x > 64 || y < 1 || y > 64) return 0.f;
    return xp[((size_t)((nimg << 12) + ((y - 1) << 6) + (x - 1))) * C_ + c];
}

__global__ void __launch_bounds__(256)
dcn_sample(const float* __restrict__ om, const float* __restrict__ xproj,
           _Float16* __restrict__ A2)
{
    const int t    = threadIdx.x;
    const int lane = t & 31;
    const int wid  = blockIdx.x * 8 + (t >> 5);
    const int m    = wid >> 3;                 // pixel row in [0, 16384)
    const int g    = wid & 7;                  // group
    const int nimg = m >> 12, hw = m & 4095, h = hw >> 6, w = hw & 63;
    const int c    = g * GC_ + lane;

    const float* row = om + (size_t)m * NOM;
    float ox[P_], oy[P_], lg[P_];
    float mx = -1e30f;
    #pragma unroll
    for (int p = 0; p < P_; p++) {
        ox[p] = row[(g * P_ + p) * 2];
        oy[p] = row[(g * P_ + p) * 2 + 1];
        lg[p] = row[NOFF + g * P_ + p];
        mx = fmaxf(mx, lg[p]);
    }
    float se = 0.f;
    #pragma unroll
    for (int p = 0; p < P_; p++) { lg[p] = __expf(lg[p] - mx); se += lg[p]; }
    const float inv = 1.f / se;

    float acc = 0.f;
    #pragma unroll
    for (int p = 0; p < P_; p++) {
        // kgrid is kw-major: kx = p/3, ky = p%3. In padded coords:
        // gx = w + kx + off_x ; gy = h + ky + off_y  (algebra folds the /66
        // normalization, *66 denormalization and the -0.5 grid_sample shift)
        float gx = (float)(w + (p / 3)) + ox[p];
        float gy = (float)(h + (p % 3)) + oy[p];
        float x0f = floorf(gx), y0f = floorf(gy);
        int   x0 = (int)x0f,   y0 = (int)y0f;
        float fx = gx - x0f,   fy = gy - y0f;
        float wgt = lg[p] * inv;
        float v00 = dcn_tap(xproj, nimg, y0,     x0,     c);
        float v10 = dcn_tap(xproj, nimg, y0,     x0 + 1, c);
        float v01 = dcn_tap(xproj, nimg, y0 + 1, x0,     c);
        float v11 = dcn_tap(xproj, nimg, y0 + 1, x0 + 1, c);
        acc += wgt * ((1.f - fx) * (1.f - fy) * v00 + fx * (1.f - fy) * v10
                    + (1.f - fx) * fy * v01 + fx * fy * v11);
    }
    A2[(size_t)m * C_ + c] = (_Float16)acc;
}

// ---------------------------------------------------------------------------
// Launch
// ---------------------------------------------------------------------------
extern "C" void kernel_launch(void* const* d_in, const int* in_sizes, int n_in,
                              void* d_out, int out_size, void* d_ws, size_t ws_size,
                              hipStream_t stream)
{
    const float* x       = (const float*)d_in[0];
    const float* w_in    = (const float*)d_in[1];
    const float* b_in    = (const float*)d_in[2];
    const float* w_dw    = (const float*)d_in[3];
    const float* b_dw    = (const float*)d_in[4];
    const float* ln_g    = (const float*)d_in[5];
    const float* ln_b    = (const float*)d_in[6];
    const float* w_off   = (const float*)d_in[7];
    const float* b_off   = (const float*)d_in[8];
    const float* w_mask  = (const float*)d_in[9];
    const float* b_mask  = (const float*)d_in[10];
    const float* w_out   = (const float*)d_in[11];
    const float* b_out   = (const float*)d_in[12];
    const float* bn_g    = (const float*)d_in[13];
    const float* bn_b    = (const float*)d_in[14];
    const float* bn_mean = (const float*)d_in[15];
    const float* bn_var  = (const float*)d_in[16];
    (void)in_sizes; (void)n_in; (void)out_size; (void)ws_size;

    char* ws = (char*)d_ws;
    size_t off = 0;
    auto alloc = [&](size_t bytes) -> void* {
        void* p = ws + off;
        off = (off + bytes + 255) & ~(size_t)255;
        return p;
    };
    _Float16* Win   = (_Float16*)alloc((size_t)C_ * C_ * 2);
    _Float16* Wout  = (_Float16*)alloc((size_t)C_ * C_ * 2);
    _Float16* Wom   = (_Float16*)alloc((size_t)C_ * NOM * 2);
    float*    bom   = (float*)   alloc((size_t)NOM * 4);
    _Float16* A0    = (_Float16*)alloc((size_t)MTOT * C_ * 2);   // x, NHWC f16
    _Float16* A1    = (_Float16*)alloc((size_t)MTOT * C_ * 2);   // GELU branch f16
    _Float16* A2    = (_Float16*)alloc((size_t)MTOT * C_ * 2);   // aggregated f16
    float*    xproj = (float*)   alloc((size_t)MTOT * C_ * 4);   // input_proj f32
    float*    om    = (float*)   alloc((size_t)MTOT * NOM * 4);  // offsets+logits

    prep_params<<<256, 256, 0, stream>>>(w_in, w_off, w_mask, b_off, b_mask,
                                         w_out, Win, Wom, Wout, bom);
    nchw_to_nhwc_f16<<<512, 256, 0, stream>>>(x, A0);

    // input_proj: xl @ w_in + b_in -> xproj (f32 NHWC)
    gemm_wmma<0><<<dim3(8, 256), 256, 0, stream>>>(
        A0, Win, b_in, xproj, MTOT, C_, nullptr, nullptr, nullptr, nullptr);

    // depthwise + LN + GELU -> A1 (f16 NHWC)
    dw_ln_gelu<<<MTOT, 256, 0, stream>>>(A0, w_dw, b_dw, ln_g, ln_b, A1);

    // offset/mask projection: A1 @ [w_off|w_mask|pad] + bias -> om (f32)
    gemm_wmma<0><<<dim3(NOM / 32, 256), 256, 0, stream>>>(
        A1, Wom, bom, om, MTOT, NOM, nullptr, nullptr, nullptr, nullptr);

    // deformable bilinear sampling + softmax mask aggregation -> A2 (f16)
    dcn_sample<<<MTOT, 256, 0, stream>>>(om, xproj, A2);

    // output_proj + bias + BN + SiLU -> d_out (f32 NCHW)
    gemm_wmma<1><<<dim3(8, 256), 256, 0, stream>>>(
        A2, Wout, b_out, (float*)d_out, MTOT, C_, bn_g, bn_b, bn_mean, bn_var);
}
